// WhisperNPUSelfAttention_63943473102884
// MI455X (gfx1250) — matrix operand order repaired and checked by reading, not compiled, not run on hardware
//
#include <hip/hip_runtime.h>

#define S_LEN   4096
#define DMODEL  512
#define NHEADS  8
#define HDIM    64

typedef __attribute__((ext_vector_type(16))) __bf16 v16bf;
typedef __attribute__((ext_vector_type(8)))  float  v8f;

union FragBF { unsigned int u[8]; v16bf v; };

__device__ __forceinline__ unsigned short f2bf(float f) {
  unsigned int u = __float_as_uint(f);
  u += 0x7FFFu + ((u >> 16) & 1u);          // round-to-nearest-even
  return (unsigned short)(u >> 16);
}

// ---- fragment loaders (per CDNA5 ISA 16-bit WMMA VGPR layouts) -------------
// A (16x32, MxK): lanes 0-15 row=lane, VGPR0..3 = K{0..7}, VGPR4..7 = K{16..23}
//                 lanes 16-31 same rows, K offset +8 (i.e. {8..15},{24..31})
__device__ __forceinline__ v16bf load_a_frag_g(const unsigned short* p,
                                               int row0, int k0, int ld) {
  const int lane = threadIdx.x & 31;
  const int row  = row0 + (lane & 15);
  const int kb   = (lane < 16) ? 0 : 8;
  const unsigned short* rp = p + row * ld + k0 + kb;
  FragBF f;
#pragma unroll
  for (int j = 0; j < 4; ++j) {
    f.u[j]     = *(const unsigned int*)(rp + 2 * j);
    f.u[j + 4] = *(const unsigned int*)(rp + 16 + 2 * j);
  }
  return f.v;
}

// B (32x16, KxN): element (k, n) = p[(col0+n)*ld + k0 + k]
// lanes 0-15: VGPR0..3 = K{0..7}, VGPR4..7 = K{16..23}; lanes 16-31: K offset +8
__device__ __forceinline__ v16bf load_b_frag(const unsigned short* p,
                                             int col0, int k0, int ld) {
  const int lane = threadIdx.x & 31;
  const int col  = col0 + (lane & 15);
  const int kh   = (lane < 16) ? 0 : 8;
  const unsigned short* cp = p + col * ld + k0 + kh;
  FragBF f;
#pragma unroll
  for (int j = 0; j < 4; ++j) {
    f.u[j]     = *(const unsigned int*)(cp + 2 * j);
    f.u[j + 4] = *(const unsigned int*)(cp + 16 + 2 * j);
  }
  return f.v;
}

// A-fragment from a 16x32 row-major bf16 tile staged in LDS
__device__ __forceinline__ v16bf load_a_frag_lds(const unsigned short* p) {
  const int lane = threadIdx.x & 31;
  const int row  = lane & 15;
  const int kb   = (lane < 16) ? 0 : 8;
  const unsigned short* rp = p + row * 32 + kb;
  FragBF f;
#pragma unroll
  for (int j = 0; j < 4; ++j) {
    f.u[j]     = *(const unsigned int*)(rp + 2 * j);
    f.u[j + 4] = *(const unsigned int*)(rp + 16 + 2 * j);
  }
  return f.v;
}

#define WMMA_BF16(a, b, c) \
  __builtin_amdgcn_wmma_f32_16x16x32_bf16(false, (a), false, (b), (short)0, (c), false, false)

// ---------------------------------------------------------------------------
__global__ void f32_to_bf16_kernel(const float* __restrict__ src,
                                   unsigned short* __restrict__ dst, int n) {
  int i = blockIdx.x * blockDim.x + threadIdx.x;
  if (i < n) dst[i] = f2bf(src[i]);
}

// One wave per 16(m)x64(n) output tile.  out[m,n] = sum_k xb[m,k]*W[n,k] + b[n]
// w==0 -> Q (scaled by 1/sqrt(hd)), w==1 -> K, w==2 -> V stored transposed Vt[n][m]
__global__ void qkv_gemm_kernel(const unsigned short* __restrict__ xb,
                                const unsigned short* __restrict__ wqb,
                                const unsigned short* __restrict__ wkb,
                                const unsigned short* __restrict__ wvb,
                                const float* __restrict__ bq,
                                const float* __restrict__ bk,
                                const float* __restrict__ bv,
                                unsigned short* __restrict__ Qb,
                                unsigned short* __restrict__ Kb,
                                unsigned short* __restrict__ Vt) {
  const int m0 = blockIdx.x * 16;
  const int n0 = blockIdx.y * 64;
  const int w  = blockIdx.z;
  const unsigned short* wb = (w == 0) ? wqb : (w == 1) ? wkb : wvb;
  const float* bias        = (w == 0) ? bq  : (w == 1) ? bk  : bv;

  v8f c[4] = {};
  for (int kk = 0; kk < DMODEL; kk += 32) {
    v16bf a = load_a_frag_g(xb, m0, kk, DMODEL);
#pragma unroll
    for (int t = 0; t < 4; ++t) {
      v16bf b = load_b_frag(wb, n0 + 16 * t, kk, DMODEL);
      c[t] = WMMA_BF16(a, b, c[t]);
    }
  }

  const int lane = threadIdx.x & 31;
  const int col  = lane & 15;
  const int half = lane >> 4;
#pragma unroll
  for (int t = 0; t < 4; ++t) {
    const int n = n0 + 16 * t + col;
    const float bn = bias[n];
#pragma unroll
    for (int j = 0; j < 8; ++j) {
      const int m = m0 + j + 8 * half;
      const float v = c[t][j] + bn;
      if (w == 0)      Qb[m * DMODEL + n] = f2bf(v * 0.125f);  // fold 1/sqrt(64)
      else if (w == 1) Kb[m * DMODEL + n] = f2bf(v);
      else             Vt[n * S_LEN + m]  = f2bf(v);           // transposed
    }
  }
}

// ---------------------------------------------------------------------------
// Flash-attention: one wave per 16-row Q block per head.  K/V chunks (32 cols)
// are double-buffered into per-wave LDS slabs with gfx1250 async-to-LDS DMA
// (ASYNCcnt-tracked), so the next chunk streams in while the current one is
// consumed by WMMA + online softmax.
#define ATTN_WAVES 2
#define K_TILE_U16 (32 * HDIM)   // 32 rows x 64 d  (4 KB)
#define V_TILE_U16 (HDIM * 32)   // 64 d-rows x 32 s (4 KB)
#define STAGE_U16  (K_TILE_U16 + V_TILE_U16)

__global__ void attn_kernel(const unsigned short* __restrict__ Qb,
                            const unsigned short* __restrict__ Kb,
                            const unsigned short* __restrict__ Vt,
                            unsigned short* __restrict__ Ob) {
  __shared__ __align__(16) unsigned short stage[ATTN_WAVES][2][STAGE_U16]; // 32 KB
  __shared__ __align__(16) unsigned short Plds[ATTN_WAVES][16 * 32];       //  2 KB
  const int widb = threadIdx.x >> 5;
  const int wid  = blockIdx.x * ATTN_WAVES + widb;    // 2048 waves total
  const int h    = wid >> 8;
  const int qb   = wid & 255;
  const int m0   = qb * 16;
  const int lane = threadIdx.x & 31;
  const int col  = lane & 15;
  const int half = lane >> 4;

  // Hoisted loop-invariant Q fragments (hd=64 -> two K=32 fragments)
  const v16bf aQ0 = load_a_frag_g(Qb, m0, h * HDIM + 0,  DMODEL);
  const v16bf aQ1 = load_a_frag_g(Qb, m0, h * HDIM + 32, DMODEL);

  // Issue one chunk's async DMA: 8x b128 for K (32 rows x 128B) and
  // 8x b128 for Vt (64 rows x 64B).  Per-lane 16B tasks, wave-uniform control.
  auto issue_chunk = [&](int n0, int buf) {
    const unsigned kl0 = (unsigned)(size_t)&stage[widb][buf][0];
    const unsigned vl0 = (unsigned)(size_t)&stage[widb][buf][K_TILE_U16];
#pragma unroll
    for (int i = 0; i < 8; ++i) {
      const int r = 4 * i + (lane >> 3);                  // K row in chunk
      unsigned long long g = (unsigned long long)(size_t)
          (Kb + (size_t)(n0 + r) * DMODEL + h * HDIM) + (unsigned)((lane & 7) * 16);
      unsigned l = kl0 + (unsigned)(r * 128 + (lane & 7) * 16);
      asm volatile("global_load_async_to_lds_b128 %0, %1, off"
                   :: "v"(l), "v"(g) : "memory");
    }
#pragma unroll
    for (int i = 0; i < 8; ++i) {
      const int d = 8 * i + (lane >> 2);                  // Vt feature row
      unsigned long long g = (unsigned long long)(size_t)
          (Vt + (size_t)(h * HDIM + d) * S_LEN + n0) + (unsigned)((lane & 3) * 16);
      unsigned l = vl0 + (unsigned)(d * 64 + (lane & 3) * 16);
      asm volatile("global_load_async_to_lds_b128 %0, %1, off"
                   :: "v"(l), "v"(g) : "memory");
    }
  };

  v8f acc[4] = {};
  float rmax[8], rsum[8];
#pragma unroll
  for (int j = 0; j < 8; ++j) { rmax[j] = -1e30f; rsum[j] = 0.0f; }

  unsigned short* myP = &Plds[widb][0];
  const int nEnd = m0 + 16;          // causal: chunks with n0 < nEnd
  int cur = 0;
  issue_chunk(0, 0);

  for (int n0 = 0; n0 < nEnd; n0 += 32) {
    const bool hasNext = (n0 + 32) < nEnd;
    if (hasNext) {
      issue_chunk(n0 + 32, cur ^ 1);                 // prefetch next chunk
      asm volatile("s_wait_asynccnt 0x10" ::: "memory");  // current 16 done
    } else {
      asm volatile("s_wait_asynccnt 0x0" ::: "memory");
    }
    const unsigned short* Kt = &stage[widb][cur][0];          // [n][d], ld=64
    const unsigned short* Vti = &stage[widb][cur][K_TILE_U16]; // [d][s], ld=32

    // scores: S = Q K^T  (two 16x16 n-tiles, two K=32 steps over hd)
    v8f s[2] = {};
#pragma unroll
    for (int tn = 0; tn < 2; ++tn) {
      v16bf b0 = load_b_frag(Kt, 16 * tn, 0, HDIM);
      s[tn] = WMMA_BF16(aQ0, b0, s[tn]);
      v16bf b1 = load_b_frag(Kt, 16 * tn, 32, HDIM);
      s[tn] = WMMA_BF16(aQ1, b1, s[tn]);
    }

    // causal mask only where the chunk crosses the diagonal
    if (n0 + 31 > m0) {
#pragma unroll
      for (int tn = 0; tn < 2; ++tn) {
        const int n = n0 + 16 * tn + col;
#pragma unroll
        for (int j = 0; j < 8; ++j) {
          const int m = m0 + j + 8 * half;
          if (n > m) s[tn][j] = -1e30f;
        }
      }
    }

    // online softmax (rows live as (VGPR j, lane-half); reduce over 16 lanes)
#pragma unroll
    for (int j = 0; j < 8; ++j) {
      float cm = fmaxf(s[0][j], s[1][j]);
      cm = fmaxf(cm, __shfl_xor(cm, 1, 32));
      cm = fmaxf(cm, __shfl_xor(cm, 2, 32));
      cm = fmaxf(cm, __shfl_xor(cm, 4, 32));
      cm = fmaxf(cm, __shfl_xor(cm, 8, 32));
      const float mnew  = fmaxf(rmax[j], cm);
      const float scale = __expf(rmax[j] - mnew);
      const float p0 = __expf(s[0][j] - mnew);
      const float p1 = __expf(s[1][j] - mnew);
      s[0][j] = p0; s[1][j] = p1;
      float cs = p0 + p1;
      cs += __shfl_xor(cs, 1, 32);
      cs += __shfl_xor(cs, 2, 32);
      cs += __shfl_xor(cs, 4, 32);
      cs += __shfl_xor(cs, 8, 32);
      rsum[j] = rsum[j] * scale + cs;
      rmax[j] = mnew;
#pragma unroll
      for (int t = 0; t < 4; ++t) acc[t][j] *= scale;
    }

    // restage P (f32 D-layout) -> LDS bf16 row-major -> A-layout fragment.
    // Per-wave private slab; DS ops are in-order within a wave.
#pragma unroll
    for (int tn = 0; tn < 2; ++tn) {
#pragma unroll
      for (int j = 0; j < 8; ++j) {
        const int r = j + 8 * half;
        myP[r * 32 + 16 * tn + col] = f2bf(s[tn][j]);
      }
    }
    const v16bf aP = load_a_frag_lds(myP);

    // acc += P @ V   (staged Vt tile is [d][s] so B k-dim runs contiguous)
#pragma unroll
    for (int dt = 0; dt < 4; ++dt) {
      v16bf bV = load_b_frag(Vti, 16 * dt, 0, 32);
      acc[dt] = WMMA_BF16(aP, bV, acc[dt]);
    }
    cur ^= 1;
  }

  // normalize and write O (bf16, row-major [s][dmodel])
#pragma unroll
  for (int j = 0; j < 8; ++j) {
    const float inv = 1.0f / rsum[j];
    const int m = m0 + j + 8 * half;
#pragma unroll
    for (int dt = 0; dt < 4; ++dt) {
      const int d = h * HDIM + 16 * dt + col;
      Ob[m * DMODEL + d] = f2bf(acc[dt][j] * inv);
    }
  }
}

// Output projection: out = O @ Wo^T + bo (f32 result)
__global__ void out_gemm_kernel(const unsigned short* __restrict__ Ob,
                                const unsigned short* __restrict__ wob,
                                const float* __restrict__ bo,
                                float* __restrict__ out) {
  const int m0 = blockIdx.x * 16;
  const int n0 = blockIdx.y * 64;
  v8f c[4] = {};
  for (int kk = 0; kk < DMODEL; kk += 32) {
    v16bf a = load_a_frag_g(Ob, m0, kk, DMODEL);
#pragma unroll
    for (int t = 0; t < 4; ++t) {
      v16bf b = load_b_frag(wob, n0 + 16 * t, kk, DMODEL);
      c[t] = WMMA_BF16(a, b, c[t]);
    }
  }
  const int lane = threadIdx.x & 31;
  const int col  = lane & 15;
  const int half = lane >> 4;
#pragma unroll
  for (int t = 0; t < 4; ++t) {
    const int n = n0 + 16 * t + col;
    const float bn = bo[n];
#pragma unroll
    for (int j = 0; j < 8; ++j) {
      const int m = m0 + j + 8 * half;
      out[m * DMODEL + n] = c[t][j] + bn;
    }
  }
}

// ---------------------------------------------------------------------------
extern "C" void kernel_launch(void* const* d_in, const int* in_sizes, int n_in,
                              void* d_out, int out_size, void* d_ws, size_t ws_size,
                              hipStream_t stream) {
  const float* x  = (const float*)d_in[0];
  const float* wq = (const float*)d_in[1];
  const float* bq = (const float*)d_in[2];
  const float* wk = (const float*)d_in[3];
  const float* bk = (const float*)d_in[4];
  const float* wv = (const float*)d_in[5];
  const float* bv = (const float*)d_in[6];
  const float* wo = (const float*)d_in[7];
  const float* bo = (const float*)d_in[8];
  float* out = (float*)d_out;

  const size_t SD = (size_t)S_LEN * DMODEL;    // 2,097,152 elems
  const size_t DD = (size_t)DMODEL * DMODEL;   //   262,144 elems

  unsigned short* ws  = (unsigned short*)d_ws; // ~22 MB of bf16 scratch total
  unsigned short* xb  = ws;
  unsigned short* wqb = xb  + SD;
  unsigned short* wkb = wqb + DD;
  unsigned short* wvb = wkb + DD;
  unsigned short* wob = wvb + DD;
  unsigned short* Qb  = wob + DD;
  unsigned short* Kb  = Qb  + SD;
  unsigned short* Vt  = Kb  + SD;   // [dmodel][seq] (per-head V transposed)
  unsigned short* Ob  = Vt  + SD;

  f32_to_bf16_kernel<<<(int)((SD + 255) / 256), 256, 0, stream>>>(x,  xb,  (int)SD);
  f32_to_bf16_kernel<<<(int)((DD + 255) / 256), 256, 0, stream>>>(wq, wqb, (int)DD);
  f32_to_bf16_kernel<<<(int)((DD + 255) / 256), 256, 0, stream>>>(wk, wkb, (int)DD);
  f32_to_bf16_kernel<<<(int)((DD + 255) / 256), 256, 0, stream>>>(wv, wvb, (int)DD);
  f32_to_bf16_kernel<<<(int)((DD + 255) / 256), 256, 0, stream>>>(wo, wob, (int)DD);

  qkv_gemm_kernel<<<dim3(S_LEN / 16, DMODEL / 64, 3), 32, 0, stream>>>(
      xb, wqb, wkb, wvb, bq, bk, bv, Qb, Kb, Vt);

  attn_kernel<<<dim3((S_LEN / 16) * NHEADS / ATTN_WAVES), 32 * ATTN_WAVES, 0, stream>>>(
      Qb, Kb, Vt, Ob);

  out_gemm_kernel<<<dim3(S_LEN / 16, DMODEL / 64), 32, 0, stream>>>(Ob, wob, bo, out);
}
